// StarNet_48180943126583
// MI455X (gfx1250) — matrix operand order repaired and checked by reading, not compiled, not run on hardware
//
#include <hip/hip_runtime.h>
#include <math.h>

typedef __attribute__((ext_vector_type(16))) _Float16 v16h;
typedef __attribute__((ext_vector_type(8)))  _Float16 v8h;
typedef __attribute__((ext_vector_type(8)))  float    v8f;

// ---------------------------------------------------------------------------
// Packing kernels: f32 -> zero-padded f16 buffers for WMMA fragments.
// ---------------------------------------------------------------------------

// A: [M][K] f32 (row stride lda) -> [Mpad][Kpad] f16, batched via blockIdx.y
__global__ void pack_a_f16(const float* __restrict__ A, _Float16* __restrict__ A16,
                           int M, int K, int lda, int Mpad, int Kpad,
                           long sStride, long dStride) {
  int b = blockIdx.y;
  long idx = (long)blockIdx.x * 256 + threadIdx.x;
  long tot = (long)Mpad * Kpad;
  if (idx >= tot) return;
  int k = (int)(idx % Kpad);
  int m = (int)(idx / Kpad);
  float v = (m < M && k < K) ? A[(long)b * sStride + (long)m * lda + k] : 0.f;
  A16[(long)b * dStride + idx] = (_Float16)v;
}

// B: [K][N] f32 -> Bt [Npad][Kpad] f16 (transposed, zero-padded)
__global__ void pack_bt_f16(const float* __restrict__ B, _Float16* __restrict__ Bt,
                            int K, int N, int Kpad, int Npad) {
  long idx = (long)blockIdx.x * 256 + threadIdx.x;
  long tot = (long)Npad * Kpad;
  if (idx >= tot) return;
  int k = (int)(idx % Kpad);
  int n = (int)(idx / Kpad);
  float v = (k < K && n < N) ? B[(long)k * N + n] : 0.f;
  Bt[idx] = (_Float16)v;
}

// ---------------------------------------------------------------------------
// Generic WMMA GEMM: C[M][ldc] (+bias, opt relu) = A16[Mpad][Kpad] * Bt16^T
// One wave per 16x16 output tile. blockDim.x == 32 (wave32).
// ---------------------------------------------------------------------------
__global__ void __launch_bounds__(32)
wmma_gemm_f16(const _Float16* __restrict__ A, const _Float16* __restrict__ Bt,
              const float* __restrict__ bias, float* __restrict__ C,
              int M, int N, int Kpad, int ldc,
              long strideA, long strideC, int relu) {
  int mt = blockIdx.x, nt = blockIdx.y, b = blockIdx.z;
  A += (long)b * strideA;
  C += (long)b * strideC;
  int lane = threadIdx.x;
  int hs   = lane >> 4;          // half-select
  int lrow = lane & 15;
  const _Float16* arow = A  + (long)(mt * 16 + lrow) * Kpad;
  const _Float16* brow = Bt + (long)(nt * 16 + lrow) * Kpad;
  v8f acc = {};
  for (int kb = 0; kb < Kpad; kb += 32) {
    if (kb + 128 < Kpad) {                 // gentle L2 prefetch down the K stream
      __builtin_prefetch(arow + kb + 128, 0, 1);
      __builtin_prefetch(brow + kb + 128, 0, 1);
    }
    // A 16x32 f16 fragment: lane<16 -> K {0..7}U{16..23}; lane>=16 -> K {8..15}U{24..31}
    v8h alo = *(const v8h*)(arow + kb + hs * 8);
    v8h ahi = *(const v8h*)(arow + kb + 16 + hs * 8);
    // B 32x16 fragment from Bt: same K order as A (lane<16 -> K {0..7}U{16..23}; lane>=16 -> K {8..15}U{24..31})
    v8h blo = *(const v8h*)(brow + kb + hs * 8);
    v8h bhi = *(const v8h*)(brow + kb + 16 + hs * 8);
    v16h a, bf;
#pragma unroll
    for (int e = 0; e < 8; ++e) {
      a[e] = alo[e];  a[e + 8] = ahi[e];
      bf[e] = blo[e]; bf[e + 8] = bhi[e];
    }
    acc = __builtin_amdgcn_wmma_f32_16x16x32_f16(false, a, false, bf, (short)0,
                                                 acc, false, false);
  }
  int gn = nt * 16 + lrow;
#pragma unroll
  for (int r = 0; r < 8; ++r) {
    int gm = mt * 16 + r + hs * 8;
    if (gm < M && gn < N) {
      float v = acc[r];
      if (bias) v += bias[gn];
      if (relu) v = fmaxf(v, 0.f);
      C[(long)gm * ldc + gn] = v;
    }
  }
}

// ---------------------------------------------------------------------------
// conv1: direct 3->64, 5x5, stride 3, pad 2, channel-swapped input, ReLU
// x: [8][3][448][448], w: [64][3][5][5], out h1: [8][64][150][150]
// ---------------------------------------------------------------------------
__global__ void conv1_direct(const float* __restrict__ x, const float* __restrict__ w,
                             const float* __restrict__ bias, float* __restrict__ h1) {
  long idx = (long)blockIdx.x * 256 + threadIdx.x;
  const long total = 8L * 64 * 150 * 150;
  if (idx >= total) return;
  int ow = (int)(idx % 150); long t = idx / 150;
  int oh = (int)(t % 150); t /= 150;
  int oc = (int)(t % 64);  int b = (int)(t / 64);
  float acc = bias[oc];
  for (int c = 0; c < 3; ++c) {
    const float* xp = x + (((long)b * 3 + c) * 448) * 448;
    const float* wp = w + ((oc * 3 + (2 - c)) * 5) * 5;   // xi = x[ch 2,1,0]
#pragma unroll
    for (int kh = 0; kh < 5; ++kh) {
      int ih = oh * 3 - 2 + kh;
      if (ih < 0 || ih >= 448) continue;
#pragma unroll
      for (int kw = 0; kw < 5; ++kw) {
        int iw = ow * 3 - 2 + kw;
        if (iw < 0 || iw >= 448) continue;
        acc += xp[(long)ih * 448 + iw] * wp[kh * 5 + kw];
      }
    }
  }
  h1[idx] = fmaxf(acc, 0.f);
}

// ---------------------------------------------------------------------------
// conv2 as implicit GEMM with WMMA: M=2500 pixels, N=32 oc, K=1600 (=50x32)
// A gathered on the fly from h1 (f32 -> f16), B from pre-packed f16 [32][1600]
// Output ft[b][oc][pix] (ReLU).
// ---------------------------------------------------------------------------
__global__ void __launch_bounds__(32)
conv2_wmma(const float* __restrict__ h1, const _Float16* __restrict__ Bt,
           const float* __restrict__ bias, float* __restrict__ ft) {
  int mt = blockIdx.x;   // 157 tiles of 16 pixels
  int nt = blockIdx.y;   // 2 tiles of 16 channels
  int b  = blockIdx.z;
  int lane = threadIdx.x, hs = lane >> 4, lrow = lane & 15;
  int m = mt * 16 + lrow;
  int oh = m / 50, ow = m % 50;
  const float* hb = h1 + (long)b * 64 * 150 * 150;
  const _Float16* brow = Bt + (long)(nt * 16 + lrow) * 1600;
  v8f acc = {};
  for (int kb = 0; kb < 1600; kb += 32) {
    v16h a, bf;
#pragma unroll
    for (int e = 0; e < 16; ++e) {
      int k = kb + (e < 8 ? hs * 8 + e : 16 + hs * 8 + (e - 8));
      int ci = k / 25, rr = k % 25;
      int kh = rr / 5, kw = rr % 5;
      int ih = oh * 3 - 2 + kh, iw = ow * 3 - 2 + kw;
      float v = 0.f;
      if (m < 2500 && ih >= 0 && ih < 150 && iw >= 0 && iw < 150)
        v = hb[((long)ci * 150 + ih) * 150 + iw];
      a[e] = (_Float16)v;
    }
    v8h blo = *(const v8h*)(brow + kb + hs * 8);
    v8h bhi = *(const v8h*)(brow + kb + 16 + hs * 8);
#pragma unroll
    for (int e = 0; e < 8; ++e) { bf[e] = blo[e]; bf[e + 8] = bhi[e]; }
    acc = __builtin_amdgcn_wmma_f32_16x16x32_f16(false, a, false, bf, (short)0,
                                                 acc, false, false);
  }
  int n = nt * 16 + lrow;
#pragma unroll
  for (int r = 0; r < 8; ++r) {
    int gm = mt * 16 + r + hs * 8;
    if (gm < 2500)
      ft[((long)b * 32 + n) * 2500 + gm] = fmaxf(acc[r] + bias[n], 0.f);
  }
}

// ---------------------------------------------------------------------------
// Hypergraph G construction per sample (exact f32, matches _knn_hypergraph_G)
// ft: [8][32][2500] -> G: [8][32][32].  One block of 1024 threads per sample.
// ---------------------------------------------------------------------------
__global__ void __launch_bounds__(1024)
hyper_g(const float* __restrict__ ft, float* __restrict__ G) {
  int b = blockIdx.x;
  const float* f = ft + (long)b * 32 * 2500;
  __shared__ float d[32][32];
  __shared__ float H[32][32];
  __shared__ float avg[32], DE[32], dvi[32];
  int i = threadIdx.x >> 5, j = threadIdx.x & 31;

  // Gram
  const float* fi = f + (long)i * 2500;
  const float* fj = f + (long)j * 2500;
  float dot = 0.f;
  for (int k = 0; k < 2500; ++k) dot += fi[k] * fj[k];
  d[i][j] = dot;
  __syncthreads();
  float d2 = d[i][i] + d[j][j] - 2.f * dot;
  float dd = sqrtf(fmaxf(d2, 0.f));
  __syncthreads();
  d[i][j] = dd;
  __syncthreads();
  float dm = fmaxf(d[i][j], d[j][i]);
  if (i == j) dm = 0.f;
  __syncthreads();
  d[i][j] = dm;
  H[i][j] = 0.f;
  __syncthreads();

  if (j == 0) {
    float s = 0.f;
    for (int c = 0; c < 32; ++c) s += d[i][c];
    avg[i] = s / 32.f;
  }
  __syncthreads();

  // per-center k=5 smallest distances (stable ties -> lowest index), scatter into H
  if (j == 0) {
    float a = avg[i];
    float denom = a * a;                 // (m_prob*avg)^2, m_prob = 1
    unsigned used = 0;
    for (int kk = 0; kk < 5; ++kk) {
      float best = 3.4e38f; int bi = 0;
      for (int c = 0; c < 32; ++c) {
        if (used & (1u << c)) continue;
        float v = d[i][c];
        if (v < best) { best = v; bi = c; }
      }
      used |= (1u << bi);
      H[bi][i] = expf(-(best * best) / denom);   // H[node][edge=center i]
    }
  }
  __syncthreads();

  if (j == 0) {
    float sc = 0.f, sr = 0.f;
    for (int r = 0; r < 32; ++r) { sc += H[r][i]; sr += H[i][r]; }
    DE[i]  = sc;            // column (edge) sums
    dvi[i] = rsqrtf(sr);    // DV^-1/2 (row/node sums)
  }
  __syncthreads();

  float g = 0.f;
  for (int c = 0; c < 32; ++c) g += H[i][c] * H[j][c] / DE[c];
  G[((long)b * 32 + i) * 32 + j] = dvi[i] * dvi[j] * g;
}

// z = relu(G @ M1)  per sample ; M1: [8][32][512]
__global__ void hg_layer1(const float* __restrict__ G, const float* __restrict__ M1,
                          float* __restrict__ z) {
  int b = blockIdx.y;
  int idx = blockIdx.x * 256 + threadIdx.x;
  if (idx >= 32 * 512) return;
  int i = idx / 512, n = idx % 512;
  const float* g = G + (long)b * 1024 + i * 32;
  const float* m = M1 + (long)b * 32 * 512;
  float s = 0.f;
  for (int k = 0; k < 32; ++k) s += g[k] * m[k * 512 + n];
  z[(long)b * 32 * 512 + idx] = fmaxf(s, 0.f);
}

// y = z @ hg_w2 + hg_b2  ; [8][32][512] x [512][32] -> [8][32][32]
__global__ void hg_layer2(const float* __restrict__ z, const float* __restrict__ W2,
                          const float* __restrict__ b2, float* __restrict__ y) {
  int b = blockIdx.y;
  int idx = blockIdx.x * 256 + threadIdx.x;
  if (idx >= 32 * 32) return;
  int i = idx / 32, j = idx % 32;
  const float* zp = z + (long)b * 32 * 512 + i * 512;
  float s = b2[j];
  for (int k = 0; k < 512; ++k) s += zp[k] * W2[k * 32 + j];
  y[(long)b * 1024 + idx] = s;
}

// outvec[b][j] = mean_i (G @ y)[i][j] + fdec[b][j]
__global__ void gmean_add(const float* __restrict__ G, const float* __restrict__ y,
                          const float* __restrict__ fdec, float* __restrict__ outvec) {
  int b = blockIdx.x, j = threadIdx.x;   // blockDim = 32
  float s = 0.f;
  for (int k = 0; k < 32; ++k) {
    float cs = 0.f;
    for (int i2 = 0; i2 < 32; ++i2) cs += G[(long)b * 1024 + i2 * 32 + k];
    s += cs * y[(long)b * 1024 + k * 32 + j];
  }
  outvec[b * 32 + j] = s / 32.f + fdec[b * 32 + j];
}

// 2-head self-attention over B=8 tokens (head dim 256)
__global__ void attention8(const float* __restrict__ q, const float* __restrict__ k,
                           const float* __restrict__ v, float* __restrict__ o) {
  int t = threadIdx.x;
  if (t >= 16) return;
  int h = t >> 3, bi = t & 7;
  const int hd = 256;
  float s[8];
  float mx = -3.4e38f;
  for (int bj = 0; bj < 8; ++bj) {
    float dot = 0.f;
    for (int dd = 0; dd < hd; ++dd)
      dot += q[bi * 512 + h * hd + dd] * k[bj * 512 + h * hd + dd];
    s[bj] = dot / 16.f;                  // sqrt(256)
    mx = fmaxf(mx, s[bj]);
  }
  float sum = 0.f;
  for (int bj = 0; bj < 8; ++bj) { s[bj] = expf(s[bj] - mx); sum += s[bj]; }
  for (int bj = 0; bj < 8; ++bj) s[bj] /= sum;
  for (int dd = 0; dd < hd; ++dd) {
    float acc = 0.f;
    for (int bj = 0; bj < 8; ++bj) acc += s[bj] * v[bj * 512 + h * hd + dd];
    o[bi * 512 + h * hd + dd] = acc;
  }
}

__global__ void copy_f(const float* __restrict__ f, float* __restrict__ out) {
  int i = blockIdx.x * 64 + threadIdx.x;
  if (i < 256) out[i] = f[i];
}

// ---------------------------------------------------------------------------
extern "C" void kernel_launch(void* const* d_in, const int* in_sizes, int n_in,
                              void* d_out, int out_size, void* d_ws, size_t ws_size,
                              hipStream_t stream) {
  const float* x       = (const float*)d_in[0];
  const float* x_f     = (const float*)d_in[1];
  const float* conv1_w = (const float*)d_in[2];
  const float* conv1_b = (const float*)d_in[3];
  const float* conv2_w = (const float*)d_in[4];
  const float* conv2_b = (const float*)d_in[5];
  const float* enc_w   = (const float*)d_in[6];
  const float* enc_b   = (const float*)d_in[7];
  const float* wq = (const float*)d_in[8];  const float* bq = (const float*)d_in[9];
  const float* wk = (const float*)d_in[10]; const float* bk = (const float*)d_in[11];
  const float* wv = (const float*)d_in[12]; const float* bv = (const float*)d_in[13];
  const float* wo = (const float*)d_in[14]; const float* bo = (const float*)d_in[15];
  const float* dec_w = (const float*)d_in[16]; const float* dec_b = (const float*)d_in[17];
  const float* hg_w1 = (const float*)d_in[18]; const float* hg_b1 = (const float*)d_in[19];
  const float* hg_w2 = (const float*)d_in[20]; const float* hg_b2 = (const float*)d_in[21];
  const float* head_w = (const float*)d_in[22]; const float* head_b = (const float*)d_in[23];
  float* out = (float*)d_out;            // [8*1000 logits][8*32 f]

  size_t off = 0;
  auto alloc = [&](size_t bytes) -> void* {
    void* p = (char*)d_ws + off;
    off += (bytes + 255) & ~(size_t)255;
    return p;
  };

  float*    h1    = (float*)   alloc(11520000ULL * 4);        // [8][64][150][150]
  float*    ft    = (float*)   alloc(640000ULL * 4);          // [8][32][2500]
  _Float16* c2bt  = (_Float16*)alloc(32ULL * 1600 * 2);       // conv2 W [32][1600]
  _Float16* ft16  = (_Float16*)alloc(8ULL * 32 * 2528 * 2);   // A for hg1, Kpad 2528
  _Float16* w1t   = (_Float16*)alloc(512ULL * 2528 * 2);      // hg_w1^T
  float*    M1    = (float*)   alloc(8ULL * 32 * 512 * 4);
  float*    Gm    = (float*)   alloc(8ULL * 1024 * 4);
  float*    zb    = (float*)   alloc(8ULL * 32 * 512 * 4);
  float*    yb    = (float*)   alloc(8ULL * 1024 * 4);
  float*    f1    = (float*)   alloc(8ULL * 512 * 4);
  float*    qb    = (float*)   alloc(8ULL * 512 * 4);
  float*    kb    = (float*)   alloc(8ULL * 512 * 4);
  float*    vb    = (float*)   alloc(8ULL * 512 * 4);
  float*    att   = (float*)   alloc(8ULL * 512 * 4);
  float*    f2    = (float*)   alloc(8ULL * 512 * 4);
  float*    fdec  = (float*)   alloc(8ULL * 32 * 4);
  float*    ovec  = (float*)   alloc(8ULL * 32 * 4);
  _Float16* a_xf  = (_Float16*)alloc(16ULL * 64 * 2);
  _Float16* a_f1  = (_Float16*)alloc(16ULL * 512 * 2);
  _Float16* a_at  = (_Float16*)alloc(16ULL * 512 * 2);
  _Float16* a_f2  = (_Float16*)alloc(16ULL * 512 * 2);
  _Float16* a_ov  = (_Float16*)alloc(16ULL * 32 * 2);
  _Float16* encbt = (_Float16*)alloc(512ULL * 64 * 2);
  _Float16* wqbt  = (_Float16*)alloc(512ULL * 512 * 2);
  _Float16* wkbt  = (_Float16*)alloc(512ULL * 512 * 2);
  _Float16* wvbt  = (_Float16*)alloc(512ULL * 512 * 2);
  _Float16* wobt  = (_Float16*)alloc(512ULL * 512 * 2);
  _Float16* decbt = (_Float16*)alloc(32ULL * 512 * 2);
  _Float16* hdbt  = (_Float16*)alloc(1008ULL * 32 * 2);
  (void)ws_size; (void)in_sizes; (void)n_in; (void)out_size;

  auto nblk = [](long tot, int bs) { return (unsigned)((tot + bs - 1) / bs); };

  // ---- conv branch ----
  conv1_direct<<<nblk(11520000L, 256), 256, 0, stream>>>(x, conv1_w, conv1_b, h1);
  pack_a_f16<<<dim3(nblk(32L * 1600, 256), 1), 256, 0, stream>>>(
      conv2_w, c2bt, 32, 1600, 1600, 32, 1600, 0, 0);
  conv2_wmma<<<dim3(157, 2, 8), 32, 0, stream>>>(h1, c2bt, conv2_b, ft);

  // ---- hypergraph branch ----
  pack_a_f16<<<dim3(nblk(32L * 2528, 256), 8), 256, 0, stream>>>(
      ft, ft16, 32, 2500, 2500, 32, 2528, 32L * 2500, 32L * 2528);
  pack_bt_f16<<<dim3(nblk(512L * 2528, 256), 1), 256, 0, stream>>>(
      hg_w1, w1t, 2500, 512, 2528, 512);
  wmma_gemm_f16<<<dim3(2, 32, 8), 32, 0, stream>>>(
      ft16, w1t, hg_b1, M1, 32, 512, 2528, 512, 32L * 2528, 32L * 512, 0);
  hyper_g<<<8, 1024, 0, stream>>>(ft, Gm);
  hg_layer1<<<dim3(64, 8), 256, 0, stream>>>(Gm, M1, zb);
  hg_layer2<<<dim3(4, 8), 256, 0, stream>>>(zb, hg_w2, hg_b2, yb);

  // ---- feature branch ----
  pack_a_f16<<<dim3(nblk(16L * 64, 256), 1), 256, 0, stream>>>(
      x_f, a_xf, 8, 36, 36, 16, 64, 0, 0);
  pack_bt_f16<<<dim3(nblk(512L * 64, 256), 1), 256, 0, stream>>>(
      enc_w, encbt, 36, 512, 64, 512);
  wmma_gemm_f16<<<dim3(1, 32, 1), 32, 0, stream>>>(
      a_xf, encbt, enc_b, f1, 8, 512, 64, 512, 0, 0, 1);

  pack_a_f16<<<dim3(nblk(16L * 512, 256), 1), 256, 0, stream>>>(
      f1, a_f1, 8, 512, 512, 16, 512, 0, 0);
  pack_bt_f16<<<dim3(nblk(512L * 512, 256), 1), 256, 0, stream>>>(wq, wqbt, 512, 512, 512, 512);
  pack_bt_f16<<<dim3(nblk(512L * 512, 256), 1), 256, 0, stream>>>(wk, wkbt, 512, 512, 512, 512);
  pack_bt_f16<<<dim3(nblk(512L * 512, 256), 1), 256, 0, stream>>>(wv, wvbt, 512, 512, 512, 512);
  wmma_gemm_f16<<<dim3(1, 32, 1), 32, 0, stream>>>(a_f1, wqbt, bq, qb, 8, 512, 512, 512, 0, 0, 0);
  wmma_gemm_f16<<<dim3(1, 32, 1), 32, 0, stream>>>(a_f1, wkbt, bk, kb, 8, 512, 512, 512, 0, 0, 0);
  wmma_gemm_f16<<<dim3(1, 32, 1), 32, 0, stream>>>(a_f1, wvbt, bv, vb, 8, 512, 512, 512, 0, 0, 0);
  attention8<<<1, 32, 0, stream>>>(qb, kb, vb, att);

  pack_a_f16<<<dim3(nblk(16L * 512, 256), 1), 256, 0, stream>>>(
      att, a_at, 8, 512, 512, 16, 512, 0, 0);
  pack_bt_f16<<<dim3(nblk(512L * 512, 256), 1), 256, 0, stream>>>(wo, wobt, 512, 512, 512, 512);
  wmma_gemm_f16<<<dim3(1, 32, 1), 32, 0, stream>>>(a_at, wobt, bo, f2, 8, 512, 512, 512, 0, 0, 0);

  pack_a_f16<<<dim3(nblk(16L * 512, 256), 1), 256, 0, stream>>>(
      f2, a_f2, 8, 512, 512, 16, 512, 0, 0);
  pack_bt_f16<<<dim3(nblk(32L * 512, 256), 1), 256, 0, stream>>>(dec_w, decbt, 512, 32, 512, 32);
  wmma_gemm_f16<<<dim3(1, 2, 1), 32, 0, stream>>>(a_f2, decbt, dec_b, fdec, 8, 32, 512, 32, 0, 0, 1);

  // ---- merge + head ----
  gmean_add<<<8, 32, 0, stream>>>(Gm, yb, fdec, ovec);
  pack_a_f16<<<dim3(nblk(16L * 32, 256), 1), 256, 0, stream>>>(
      ovec, a_ov, 8, 32, 32, 16, 32, 0, 0);
  pack_bt_f16<<<dim3(nblk(1008L * 32, 256), 1), 256, 0, stream>>>(
      head_w, hdbt, 32, 1000, 32, 1008);
  wmma_gemm_f16<<<dim3(1, 63, 1), 32, 0, stream>>>(
      a_ov, hdbt, head_b, out, 8, 1000, 32, 1000, 0, 0, 0);
  copy_f<<<4, 64, 0, stream>>>(fdec, out + 8000);
}
